// TransformerDecoder_61177514165042
// MI455X (gfx1250) — matrix operand order repaired and checked by reading, not compiled, not run on hardware
//
#include <hip/hip_runtime.h>

// ---------------------------------------------------------------------------
// Transformer decoder forward for MI455X (gfx1250, wave32, WMMA bf16).
// Matmuls use v_wmma_f32_16x16x32_bf16 with f32 accumulation.
// ---------------------------------------------------------------------------

typedef __attribute__((ext_vector_type(16))) __bf16 v16bf;
typedef __attribute__((ext_vector_type(8)))  float  v8f;

namespace cfg {
constexpr int TV  = 32000;
constexpr int TD  = 1024;
constexpr int TL  = 6;
constexpr int TH  = 16;
constexpr int TFF = 4096;
constexpr int TS  = 2048;
constexpr int TB  = 2;
constexpr int THD = 64;
constexpr int TM  = TB * TS;   // 4096 token rows
}

union Frag {            // one WMMA 16x16x32 bf16 operand (32 bytes / lane)
    uint4          q[2];
    v16bf          v;
    unsigned short u[16];
};
union Pack16 { uint4 q[2]; unsigned short u[16]; };
union Pack32 { uint4 q[4]; unsigned short u[32]; };

__device__ __forceinline__ unsigned short f2bf(float f) {
    unsigned int u = __float_as_uint(f);
    u = (u + 0x7FFFu + ((u >> 16) & 1u)) >> 16;       // round-to-nearest-even
    return (unsigned short)u;
}
__device__ __forceinline__ float bf2f(unsigned short h) {
    return __uint_as_float(((unsigned int)h) << 16);
}
__device__ __forceinline__ v8f zero8() {
    v8f z;
#pragma unroll
    for (int i = 0; i < 8; ++i) z[i] = 0.0f;
    return z;
}

// ---------------------------------------------------------------------------
// fp32 -> bf16 bulk convert (n must be a multiple of 4; all sizes here are)
// ---------------------------------------------------------------------------
__global__ void cvt_bf16_kernel(const float* __restrict__ in,
                                unsigned short* __restrict__ out, size_t n4) {
    size_t i = (size_t)blockIdx.x * blockDim.x + threadIdx.x;
    if (i >= n4) return;
    float4 f = ((const float4*)in)[i];
    ushort4 o;
    o.x = f2bf(f.x); o.y = f2bf(f.y); o.z = f2bf(f.z); o.w = f2bf(f.w);
    ((ushort4*)out)[i] = o;
}

// ---------------------------------------------------------------------------
// Embedding lookup + sinusoidal positional encoding.
// One block per token row; 256 threads x 4 elements (D = 1024).
// ---------------------------------------------------------------------------
__global__ void embed_kernel(const int* __restrict__ tokens,
                             const float* __restrict__ emb,
                             float* __restrict__ x,
                             unsigned short* __restrict__ xh) {
    using namespace cfg;
    const int row = blockIdx.x;                // b*S + s
    const int pos = row & (TS - 1);
    const int tok = tokens[row];
    const int tid = threadIdx.x;
    float4 o;
    float* op = &o.x;
    const float4 ev = ((const float4*)(emb + (size_t)tok * TD))[tid];
    const float* ep = &ev.x;
#pragma unroll
    for (int j = 0; j < 4; ++j) {
        int d = tid * 4 + j;
        float div = __expf((float)(d & ~1) * (-9.210340371976184f / (float)TD));
        float ang = (float)pos * div;
        float pe  = (d & 1) ? __cosf(ang) : __sinf(ang);
        op[j] = ep[j] + pe;
    }
    ((float4*)(x + (size_t)row * TD))[tid] = o;
    ushort4 oh;
    oh.x = f2bf(o.x); oh.y = f2bf(o.y); oh.z = f2bf(o.z); oh.w = f2bf(o.w);
    ((ushort4*)(xh + (size_t)row * TD))[tid] = oh;
}

// ---------------------------------------------------------------------------
// bf16 WMMA GEMM:  C[M,N] = A[M,K] @ W[K,N] + bias  (optional ReLU)
// Block tile 64x64, 128 threads = 4 waves, each wave computes 32x32
// (2x2 grid of 16x16x32 WMMA accumulators).  K, N multiples of 32/64.
// A staged row-major in LDS (stride 40 keeps b128 alignment),
// W staged transposed so B-fragments read as contiguous b128 pairs.
// ---------------------------------------------------------------------------
template <bool RELU, bool OUTBF16>
__global__ void __launch_bounds__(128)
gemm_bf16_kernel(const unsigned short* __restrict__ A,
                 const unsigned short* __restrict__ W,
                 const float* __restrict__ bias,
                 void* __restrict__ Cout, int N, int K) {
    __shared__ unsigned short As[64 * 40];
    __shared__ unsigned short Bst[64 * 40];

    const int tid  = threadIdx.x;
    const int lane = tid & 31, wv = tid >> 5;
    const int wm = wv >> 1, wn = wv & 1;
    const int half = lane >> 4, r16 = lane & 15;
    const int m0 = blockIdx.y * 64, n0 = blockIdx.x * 64;

    const int arow = tid >> 1, acol = (tid & 1) << 4;   // 64x32 A tile
    const int brow = tid >> 2, bcol = (tid & 3) << 4;   // 32x64 W tile

    v8f acc[2][2];
#pragma unroll
    for (int f = 0; f < 2; ++f)
#pragma unroll
        for (int g = 0; g < 2; ++g) acc[f][g] = zero8();

    const int ksteps = K >> 5;
    for (int kt = 0; kt < ksteps; ++kt) {
        const int k0 = kt << 5;
        const uint4* ag = (const uint4*)(A + (size_t)(m0 + arow) * K + k0 + acol);
        uint4 a0 = ag[0], a1 = ag[1];
        const uint4* bg = (const uint4*)(W + (size_t)(k0 + brow) * N + n0 + bcol);
        Pack16 bw;
        bw.q[0] = bg[0]; bw.q[1] = bg[1];

        __syncthreads();                         // previous tile fully consumed
        *(uint4*)(As + arow * 40 + acol)     = a0;
        *(uint4*)(As + arow * 40 + acol + 8) = a1;
#pragma unroll
        for (int j = 0; j < 16; ++j)             // transpose W tile into LDS
            Bst[(bcol + j) * 40 + brow] = bw.u[j];
        __syncthreads();

#pragma unroll
        for (int f = 0; f < 2; ++f) {
            Frag af;
            const unsigned short* ap = As + (wm * 32 + f * 16 + r16) * 40;
            af.q[0] = *(const uint4*)(ap + half * 8);
            af.q[1] = *(const uint4*)(ap + 16 + half * 8);
#pragma unroll
            for (int g = 0; g < 2; ++g) {
                Frag bf;
                const unsigned short* bp =
                    Bst + (wn * 32 + g * 16 + r16) * 40;
                bf.q[0] = *(const uint4*)(bp + half * 8);
                bf.q[1] = *(const uint4*)(bp + 16 + half * 8);
                acc[f][g] = __builtin_amdgcn_wmma_f32_16x16x32_bf16(
                    false, af.v, false, bf.v, (short)0, acc[f][g], false, false);
            }
        }
    }

#pragma unroll
    for (int f = 0; f < 2; ++f)
#pragma unroll
        for (int g = 0; g < 2; ++g) {
            const int col = n0 + wn * 32 + g * 16 + r16;
            const float bv = bias ? bias[col] : 0.0f;
#pragma unroll
            for (int i = 0; i < 8; ++i) {
                const int row = m0 + wm * 32 + f * 16 + i + half * 8;
                float v = acc[f][g][i] + bv;
                if (RELU) v = fmaxf(v, 0.0f);
                if (OUTBF16)
                    ((unsigned short*)Cout)[(size_t)row * N + col] = f2bf(v);
                else
                    ((float*)Cout)[(size_t)row * N + col] = v;
            }
        }
}

// ---------------------------------------------------------------------------
// Split packed QKV [B*S, 3, H, HD] into Q/K/V as [B, H, S, HD] (bf16).
// ---------------------------------------------------------------------------
__global__ void qkv_permute_kernel(const unsigned short* __restrict__ qkv,
                                   unsigned short* __restrict__ q,
                                   unsigned short* __restrict__ k,
                                   unsigned short* __restrict__ v) {
    using namespace cfg;
    const int idx = blockIdx.x * blockDim.x + threadIdx.x;  // B*S*H*HD threads
    const int hd = idx & (THD - 1);
    int t = idx >> 6;
    const int h = t & (TH - 1);  t >>= 4;
    const int s = t & (TS - 1);
    const int b = t >> 11;
    const size_t src = (size_t)(b * TS + s) * (3 * TD) + h * THD + hd;
    const size_t dst = (((size_t)b * TH + h) * TS + s) * THD + hd;
    q[dst] = qkv[src];
    k[dst] = qkv[src + TD];
    v[dst] = qkv[src + 2 * TD];
}

// ---------------------------------------------------------------------------
// Causal flash attention, bf16 WMMA, HD = 64.
// Grid (S/64, H, B); 128 threads = 4 waves; wave w owns query rows w*16..+15.
// Scores: Q(16x32 A-frags) x K-block (LDS, natural layout == B-frag layout).
// P x V:  P via LDS round trip (C-layout -> A-layout), V staged transposed.
// Online softmax with per-row running max/sum; row stats reduced across the
// 16-lane half owning that row via __shfl_xor (wave32).
// ---------------------------------------------------------------------------
__global__ void __launch_bounds__(128)
flash_attn_kernel(const unsigned short* __restrict__ Q,
                  const unsigned short* __restrict__ K,
                  const unsigned short* __restrict__ V,
                  unsigned short* __restrict__ O) {
    using namespace cfg;
    __shared__ unsigned short Ks[64 * 72];   // [key][hd]
    __shared__ unsigned short Vst[64 * 72];  // [hd][key]
    __shared__ unsigned short Ps[64 * 72];   // [q][key]

    const int tid  = threadIdx.x;
    const int lane = tid & 31, wv = tid >> 5;
    const int half = lane >> 4, r16 = lane & 15;
    const int qb = blockIdx.x * 64;
    const int h  = blockIdx.y;
    const int b  = blockIdx.z;
    const size_t bh = ((size_t)b * TH + h) * TS;

    // Q fragments straight from global (rows are contiguous HD=64 vectors).
    Frag qf[2];
    {
        const unsigned short* qp = Q + (bh + qb + wv * 16 + r16) * THD;
        qf[0].q[0] = *(const uint4*)(qp + half * 8);
        qf[0].q[1] = *(const uint4*)(qp + 16 + half * 8);
        qf[1].q[0] = *(const uint4*)(qp + 32 + half * 8);
        qf[1].q[1] = *(const uint4*)(qp + 48 + half * 8);
    }

    v8f oacc[4];
#pragma unroll
    for (int t = 0; t < 4; ++t) oacc[t] = zero8();
    float mrow[8], lrow[8];
#pragma unroll
    for (int i = 0; i < 8; ++i) { mrow[i] = -3.0e38f; lrow[i] = 0.0f; }

    const float scale = 0.125f;               // 1/sqrt(64)
    const int krow = tid >> 1, kcol = (tid & 1) << 5;
    const int rbase = qb + wv * 16 + half * 8;
    const int nkb = blockIdx.x + 1;           // causal: key blocks 0..qb/64

    for (int kb = 0; kb < nkb; ++kb) {
        // -------- stage K (raw) and V (transposed) blocks ------------------
        const uint4* kg = (const uint4*)(K + (bh + kb * 64 + krow) * THD + kcol);
        uint4 k0 = kg[0], k1 = kg[1], k2 = kg[2], k3 = kg[3];
        const uint4* vg = (const uint4*)(V + (bh + kb * 64 + krow) * THD + kcol);
        Pack32 vw;
        vw.q[0] = vg[0]; vw.q[1] = vg[1]; vw.q[2] = vg[2]; vw.q[3] = vg[3];

        __syncthreads();
        uint4* kd = (uint4*)(Ks + krow * 72 + kcol);
        kd[0] = k0; kd[1] = k1; kd[2] = k2; kd[3] = k3;
#pragma unroll
        for (int j = 0; j < 32; ++j)
            Vst[(kcol + j) * 72 + krow] = vw.u[j];
        __syncthreads();

        // -------- scores: 16q x 64keys per wave ----------------------------
        v8f sacc[4];
#pragma unroll
        for (int g = 0; g < 4; ++g) sacc[g] = zero8();
#pragma unroll
        for (int kc = 0; kc < 2; ++kc) {
#pragma unroll
            for (int g = 0; g < 4; ++g) {
                Frag bf;
                const unsigned short* bp =
                    Ks + (g * 16 + r16) * 72 + kc * 32;
                bf.q[0] = *(const uint4*)(bp + half * 8);
                bf.q[1] = *(const uint4*)(bp + 16 + half * 8);
                sacc[g] = __builtin_amdgcn_wmma_f32_16x16x32_bf16(
                    false, qf[kc].v, false, bf.v, (short)0, sacc[g], false, false);
            }
        }

        // -------- scale + causal mask --------------------------------------
        float sval[4][8];
#pragma unroll
        for (int g = 0; g < 4; ++g) {
            const int col = kb * 64 + g * 16 + r16;
#pragma unroll
            for (int i = 0; i < 8; ++i) {
                const int row = rbase + i;
                float vs = sacc[g][i] * scale;
                sval[g][i] = (col <= row) ? vs : -1.0e30f;
            }
        }

        // -------- online softmax update ------------------------------------
        float mnew[8], lsum[8], corr[8];
#pragma unroll
        for (int i = 0; i < 8; ++i) {
            float m = fmaxf(fmaxf(sval[0][i], sval[1][i]),
                            fmaxf(sval[2][i], sval[3][i]));
            m = fmaxf(m, __shfl_xor(m, 1, 32));
            m = fmaxf(m, __shfl_xor(m, 2, 32));
            m = fmaxf(m, __shfl_xor(m, 4, 32));
            m = fmaxf(m, __shfl_xor(m, 8, 32));
            mnew[i] = fmaxf(mrow[i], m);
        }
#pragma unroll
        for (int g = 0; g < 4; ++g)
#pragma unroll
            for (int i = 0; i < 8; ++i)
                sval[g][i] = __expf(sval[g][i] - mnew[i]);
#pragma unroll
        for (int i = 0; i < 8; ++i) {
            float s = sval[0][i] + sval[1][i] + sval[2][i] + sval[3][i];
            s += __shfl_xor(s, 1, 32);
            s += __shfl_xor(s, 2, 32);
            s += __shfl_xor(s, 4, 32);
            s += __shfl_xor(s, 8, 32);
            lsum[i] = s;
            corr[i] = __expf(mrow[i] - mnew[i]);
            lrow[i] = lrow[i] * corr[i] + s;
            mrow[i] = mnew[i];
        }
#pragma unroll
        for (int t = 0; t < 4; ++t)
#pragma unroll
            for (int i = 0; i < 8; ++i) oacc[t][i] *= corr[i];

        // -------- P: C-layout -> LDS -> A-layout ---------------------------
#pragma unroll
        for (int g = 0; g < 4; ++g)
#pragma unroll
            for (int i = 0; i < 8; ++i)
                Ps[(wv * 16 + half * 8 + i) * 72 + g * 16 + r16] =
                    f2bf(sval[g][i]);
        __syncthreads();   // uniform; also orders the mixed-type LDS accesses

        // -------- O += P @ V -----------------------------------------------
#pragma unroll
        for (int kc2 = 0; kc2 < 2; ++kc2) {
            Frag pf;
            const unsigned short* pp = Ps + (wv * 16 + r16) * 72 + kc2 * 32;
            pf.q[0] = *(const uint4*)(pp + half * 8);
            pf.q[1] = *(const uint4*)(pp + 16 + half * 8);
#pragma unroll
            for (int t = 0; t < 4; ++t) {
                Frag vf;
                const unsigned short* vp =
                    Vst + (t * 16 + r16) * 72 + kc2 * 32;
                vf.q[0] = *(const uint4*)(vp + half * 8);
                vf.q[1] = *(const uint4*)(vp + 16 + half * 8);
                oacc[t] = __builtin_amdgcn_wmma_f32_16x16x32_bf16(
                    false, pf.v, false, vf.v, (short)0, oacc[t], false, false);
            }
        }
    }

    // -------- normalize + write O as [B, S, H, HD] -------------------------
#pragma unroll
    for (int t = 0; t < 4; ++t)
#pragma unroll
        for (int i = 0; i < 8; ++i) {
            const int row = rbase + i;
            const float ov = oacc[t][i] / lrow[i];
            O[((size_t)(b * TS + row) * TH + h) * THD + t * 16 + r16] = f2bf(ov);
        }
}

// ---------------------------------------------------------------------------
// x = LayerNorm(x + delta) * g + b ; also refresh bf16 mirror.
// One block per row, 256 threads x 4 elements, LDS tree reduction.
// ---------------------------------------------------------------------------
__global__ void __launch_bounds__(256)
ln_kernel(float* __restrict__ x, const float* __restrict__ delta,
          const float* __restrict__ g, const float* __restrict__ bta,
          unsigned short* __restrict__ xh) {
    using namespace cfg;
    __shared__ float red[256];
    const int row = blockIdx.x, tid = threadIdx.x;
    float* xr = x + (size_t)row * TD;
    const float4 xv = ((const float4*)xr)[tid];
    const float4 dv = ((const float4*)(delta + (size_t)row * TD))[tid];
    float t[4] = {xv.x + dv.x, xv.y + dv.y, xv.z + dv.z, xv.w + dv.w};

    red[tid] = t[0] + t[1] + t[2] + t[3];
    __syncthreads();
    for (int st = 128; st > 0; st >>= 1) {
        if (tid < st) red[tid] += red[tid + st];
        __syncthreads();
    }
    const float mu = red[0] * (1.0f / (float)TD);
    __syncthreads();

    float d0 = t[0] - mu, d1 = t[1] - mu, d2 = t[2] - mu, d3 = t[3] - mu;
    red[tid] = d0 * d0 + d1 * d1 + d2 * d2 + d3 * d3;
    __syncthreads();
    for (int st = 128; st > 0; st >>= 1) {
        if (tid < st) red[tid] += red[tid + st];
        __syncthreads();
    }
    const float rs = rsqrtf(red[0] * (1.0f / (float)TD) + 1e-5f);

    const float4 gv = ((const float4*)g)[tid];
    const float4 bv = ((const float4*)bta)[tid];
    float4 y;
    y.x = d0 * rs * gv.x + bv.x;
    y.y = d1 * rs * gv.y + bv.y;
    y.z = d2 * rs * gv.z + bv.z;
    y.w = d3 * rs * gv.w + bv.w;
    ((float4*)xr)[tid] = y;
    ushort4 oh;
    oh.x = f2bf(y.x); oh.y = f2bf(y.y); oh.z = f2bf(y.z); oh.w = f2bf(y.w);
    ((ushort4*)(xh + (size_t)row * TD))[tid] = oh;
}

// ---------------------------------------------------------------------------
// Host launcher
// ---------------------------------------------------------------------------
extern "C" void kernel_launch(void* const* d_in, const int* in_sizes, int n_in,
                              void* d_out, int out_size, void* d_ws,
                              size_t ws_size, hipStream_t stream) {
    using namespace cfg;
    (void)in_sizes; (void)n_in; (void)out_size; (void)ws_size;

    const int*   tokens = (const int*)d_in[0];
    const float* emb    = (const float*)d_in[1];
    const float* qkv_w  = (const float*)d_in[2];
    const float* qkv_b  = (const float*)d_in[3];
    const float* out_w  = (const float*)d_in[4];
    const float* out_b  = (const float*)d_in[5];
    const float* fc1_w  = (const float*)d_in[6];
    const float* fc1_b  = (const float*)d_in[7];
    const float* fc2_w  = (const float*)d_in[8];
    const float* fc2_b  = (const float*)d_in[9];
    const float* ln1_g  = (const float*)d_in[10];
    const float* ln1_b  = (const float*)d_in[11];
    const float* ln2_g  = (const float*)d_in[12];
    const float* ln2_b  = (const float*)d_in[13];
    const float* w_out  = (const float*)d_in[14];
    const float* b_out  = (const float*)d_in[15];

    char* p = (char*)d_ws;
    auto alloc = [&](size_t bytes) -> void* {
        void* r = (void*)p;
        p += (bytes + 255) & ~(size_t)255;
        return r;
    };
    typedef unsigned short us;

    const size_t n_qkvw = (size_t)TL * TD * 3 * TD;
    const size_t n_outw = (size_t)TL * TD * TD;
    const size_t n_fc1w = (size_t)TL * TD * TFF;
    const size_t n_fc2w = (size_t)TL * TFF * TD;
    const size_t n_wout = (size_t)TD * TV;

    us* qkv_w_h = (us*)alloc(n_qkvw * 2);
    us* out_w_h = (us*)alloc(n_outw * 2);
    us* fc1_w_h = (us*)alloc(n_fc1w * 2);
    us* fc2_w_h = (us*)alloc(n_fc2w * 2);
    us* w_out_h = (us*)alloc(n_wout * 2);

    float* x     = (float*)alloc((size_t)TM * TD * 4);
    us*    xh    = (us*)alloc((size_t)TM * TD * 2);
    us*    qkvh  = (us*)alloc((size_t)TM * 3 * TD * 2);
    us*    qh    = (us*)alloc((size_t)TM * TD * 2);
    us*    kh    = (us*)alloc((size_t)TM * TD * 2);
    us*    vh    = (us*)alloc((size_t)TM * TD * 2);
    us*    attno = (us*)alloc((size_t)TM * TD * 2);
    float* of32  = (float*)alloc((size_t)TM * TD * 4);
    us*    hbuf  = (us*)alloc((size_t)TM * TFF * 2);

    auto cvt = [&](const float* src, us* dst, size_t n) {
        const size_t n4 = n / 4;
        cvt_bf16_kernel<<<(unsigned)((n4 + 255) / 256), 256, 0, stream>>>(src, dst, n4);
    };
    cvt(qkv_w, qkv_w_h, n_qkvw);
    cvt(out_w, out_w_h, n_outw);
    cvt(fc1_w, fc1_w_h, n_fc1w);
    cvt(fc2_w, fc2_w_h, n_fc2w);
    cvt(w_out, w_out_h, n_wout);

    embed_kernel<<<TM, 256, 0, stream>>>(tokens, emb, x, xh);

    const dim3 gemm_blk(128);
    for (int l = 0; l < TL; ++l) {
        // QKV projection -> bf16 [M, 3D]
        gemm_bf16_kernel<false, true>
            <<<dim3(3 * TD / 64, TM / 64), gemm_blk, 0, stream>>>(
                xh, qkv_w_h + (size_t)l * TD * 3 * TD, qkv_b + (size_t)l * 3 * TD,
                qkvh, 3 * TD, TD);

        qkv_permute_kernel<<<(TB * TS * TH * THD) / 256, 256, 0, stream>>>(
            qkvh, qh, kh, vh);

        flash_attn_kernel<<<dim3(TS / 64, TH, TB), 128, 0, stream>>>(
            qh, kh, vh, attno);

        // attention output projection -> f32 delta
        gemm_bf16_kernel<false, false>
            <<<dim3(TD / 64, TM / 64), gemm_blk, 0, stream>>>(
                attno, out_w_h + (size_t)l * TD * TD, out_b + (size_t)l * TD,
                of32, TD, TD);

        ln_kernel<<<TM, 256, 0, stream>>>(x, of32, ln1_g + (size_t)l * TD,
                                          ln1_b + (size_t)l * TD, xh);

        // FC1 + ReLU -> bf16 [M, FF]
        gemm_bf16_kernel<true, true>
            <<<dim3(TFF / 64, TM / 64), gemm_blk, 0, stream>>>(
                xh, fc1_w_h + (size_t)l * TD * TFF, fc1_b + (size_t)l * TFF,
                hbuf, TFF, TD);

        // FC2 -> f32 delta
        gemm_bf16_kernel<false, false>
            <<<dim3(TD / 64, TM / 64), gemm_blk, 0, stream>>>(
                hbuf, fc2_w_h + (size_t)l * TFF * TD, fc2_b + (size_t)l * TD,
                of32, TD, TFF);

        ln_kernel<<<TM, 256, 0, stream>>>(x, of32, ln2_g + (size_t)l * TD,
                                          ln2_b + (size_t)l * TD, xh);
    }

    // Final vocab projection -> f32 logits [M, V]
    gemm_bf16_kernel<false, false>
        <<<dim3(TV / 64, TM / 64), gemm_blk, 0, stream>>>(
            xh, w_out_h, b_out, (float*)d_out, TV, TD);
}
